// VectorQuantizer_5609227288994
// MI455X (gfx1250) — matrix operand order repaired and checked by reading, not compiled, not run on hardware
//
#include <hip/hip_runtime.h>

// ---------------------------------------------------------------------------
// VQ forward for MI455X (gfx1250, wave32).
// Distance rank key: ||e_k||^2 - 2*(x . e_k)  (||x||^2 row-constant, dropped).
// GEMM in split-precision bf16 on V_WMMA_F32_16X16X32_BF16:
//   x = x_hi + x_lo (bf16 RNE + residual), x.e ~= hi.hi + hi.lo + lo.hi
// Three independent accumulators for WMMA ILP; B tiles staged into LDS with
// GLOBAL_LOAD_ASYNC_TO_LDS_B128 (ASYNCcnt) when the toolchain exposes it.
// ---------------------------------------------------------------------------

typedef __attribute__((ext_vector_type(16))) __bf16        v16bf;
typedef __attribute__((ext_vector_type(8)))  float         v8f;
typedef __attribute__((ext_vector_type(4)))  float         v4f;
typedef __attribute__((ext_vector_type(4)))  unsigned int  v4u;
typedef __attribute__((ext_vector_type(4)))  int           v4i;

#define VQ_D            256
#define VQ_K            1024
#define VQ_N            65536          // BATCH * NODE = 8192 * 8
#define ROWS_PER_WAVE   16
#define ROWS_PER_BLOCK  128            // 8 waves * 16 rows
#define AS              260            // f32 LDS stride for A staging (padded)
#define BSTR            264            // bf16 LDS stride per code row (16B aligned)
#define BPLANE          (16 * BSTR)    // one plane (hi or lo) of a 16-code tile
#define BBUF            (2 * BPLANE)   // hi + lo planes
#define KTILES          (VQ_K / 16)    // 64
#define KSTEPS          (VQ_D / 32)    // 8 bf16 WMMA k-steps per 16x16 tile

#if defined(__has_builtin)
#if __has_builtin(__builtin_amdgcn_global_load_async_to_lds_b128) && \
    __has_builtin(__builtin_amdgcn_s_wait_asynccnt)
#define VQ_ASYNC_STAGE 1
#endif
#endif
#ifndef VQ_ASYNC_STAGE
#define VQ_ASYNC_STAGE 0
#endif

__device__ __forceinline__ unsigned short f32_to_bf16_rne(float f) {
    unsigned int u = __float_as_uint(f);
    u += 0x7fffu + ((u >> 16) & 1u);
    return (unsigned short)(u >> 16);
}
__device__ __forceinline__ float bf16_to_f32(unsigned short h) {
    return __uint_as_float(((unsigned int)h) << 16);
}

union BfFrag {                 // all members trivial (clang ext vectors only)
    v16bf          v;
    unsigned short s[16];
    v4u            q[2];
};

// ---------------------------------------------------------------------------
// Kernel 0: embedding -> bf16 hi/lo planes + ||e_k||^2. One block per code.
// ---------------------------------------------------------------------------
__global__ __launch_bounds__(256) void vq_prep_kernel(
    const float* __restrict__ emb,
    unsigned short* __restrict__ ehi, unsigned short* __restrict__ elo,
    float* __restrict__ enorm)
{
    __shared__ float red[256];
    const int k = blockIdx.x;       // code 0..1023
    const int d = threadIdx.x;      // dim  0..255
    const float x = emb[(size_t)k * VQ_D + d];
    const unsigned short h = f32_to_bf16_rne(x);
    ehi[(size_t)k * VQ_D + d] = h;
    elo[(size_t)k * VQ_D + d] = f32_to_bf16_rne(x - bf16_to_f32(h));
    red[d] = x * x;
    __syncthreads();
#pragma unroll
    for (int st = 128; st > 0; st >>= 1) {
        if (d < st) red[d] += red[d + st];
        __syncthreads();
    }
    if (d == 0) enorm[k] = red[0];
}

// ---------------------------------------------------------------------------
// Kernel 1: per-row argmin over 1024 codes, split-precision bf16 WMMA.
// ---------------------------------------------------------------------------
__global__ __launch_bounds__(256) void vq_argmin_kernel(
    const float* __restrict__ lat,            // [N, 256] f32
    const unsigned short* __restrict__ ehi,   // [1024, 256] bf16 bits
    const unsigned short* __restrict__ elo,   // [1024, 256] bf16 bits
    const float* __restrict__ enorm,          // [1024]
    int* __restrict__ idx_out)                // [N]
{
    extern __shared__ unsigned char smem_raw[];
    float*          As_f = (float*)smem_raw;          // phase 1: 128 x 260 f32
    unsigned short* Bs   = (unsigned short*)smem_raw; // phase 2: 2 x BBUF bf16

    const int tid  = threadIdx.x;
    const int lane = tid & 31;
    const int wave = tid >> 5;
    const int half = lane >> 4;    // 0: lanes 0-15, 1: lanes 16-31
    const int l15  = lane & 15;
    const int block_row = blockIdx.x * ROWS_PER_BLOCK;

    // ---- Phase 1: stage A block f32, coalesced ----
    for (int i = tid; i < ROWS_PER_BLOCK * (VQ_D / 4); i += 256) {
        int r = i >> 6;            // / (256/4)
        int c = (i & 63) * 4;
        v4f v = *(const v4f*)(lat + (size_t)(block_row + r) * VQ_D + c);
        float* dst = As_f + r * AS + c;
        dst[0] = v.x; dst[1] = v.y; dst[2] = v.z; dst[3] = v.w;
    }
    __syncthreads();

    // ---- Hoist + split this wave's A panel into bf16 hi/lo register frags.
    // ISA 16-bit A 16x32 layout: lane row m = lane&15;
    //   elem j   (j=0..7) = A[m][32s +      8*half + j]
    //   elem 8+j (j=0..7) = A[m][32s + 16 + 8*half + j]
    BfFrag fahi[KSTEPS], falo[KSTEPS];
    {
        const float* base = As_f + (wave * ROWS_PER_WAVE + l15) * AS + 8 * half;
#pragma unroll
        for (int s = 0; s < KSTEPS; ++s) {
#pragma unroll
            for (int j = 0; j < 8; ++j) {
                float x0 = base[32 * s + j];
                float x1 = base[32 * s + 16 + j];
                unsigned short h0 = f32_to_bf16_rne(x0);
                unsigned short h1 = f32_to_bf16_rne(x1);
                fahi[s].s[j]     = h0;
                fahi[s].s[8 + j] = h1;
                falo[s].s[j]     = f32_to_bf16_rne(x0 - bf16_to_f32(h0));
                falo[s].s[8 + j] = f32_to_bf16_rne(x1 - bf16_to_f32(h1));
            }
        }
    }
    __syncthreads();   // done with As_f; B buffers may overwrite the region

    // Stage one 16-code embedding tile (hi+lo planes) into an LDS buffer.
    // 1024 chunks of 8 bf16 (16B each): 2 planes * 16 codes * 32 chunks.
    auto stage_b = [&](int kt, unsigned short* buf) {
        for (int c = tid; c < 1024; c += 256) {
            int plane = c >> 9;
            int code  = (c >> 5) & 15;
            int col   = (c & 31) * 8;
            const unsigned short* src =
                (plane ? elo : ehi) + (size_t)(kt * 16 + code) * VQ_D + col;
            unsigned short* dst = buf + plane * BPLANE + code * BSTR + col;
#if VQ_ASYNC_STAGE
            __builtin_amdgcn_global_load_async_to_lds_b128(
                (__attribute__((address_space(1))) v4i*)src,
                (__attribute__((address_space(3))) v4i*)dst,
                /*offset=*/0, /*cpol=*/0);
#else
            *(v4u*)dst = *(const v4u*)src;
#endif
        }
    };

    stage_b(0, Bs);
#if VQ_ASYNC_STAGE
    __builtin_amdgcn_s_wait_asynccnt(0);
#endif
    __syncthreads();

    float best[8];
    int   bidx[8];
#pragma unroll
    for (int r = 0; r < 8; ++r) { best[r] = 3.4e38f; bidx[r] = 0; }

    for (int kt = 0; kt < KTILES; ++kt) {
        unsigned short* Bcur = Bs + (kt & 1) * BBUF;
        unsigned short* Bnxt = Bs + ((kt + 1) & 1) * BBUF;

        if (kt + 1 < KTILES) stage_b(kt + 1, Bnxt);
        if (kt + 2 < KTILES) {  // L2 prefetch hint two tiles ahead
            __builtin_prefetch(ehi + (size_t)(kt + 2) * 16 * VQ_D + tid * 16, 0, 1);
            __builtin_prefetch(elo + (size_t)(kt + 2) * 16 * VQ_D + tid * 16, 0, 1);
        }

        // 16x16 score tile: 8 K-steps, 3 bf16 WMMAs each, into 3 independent
        // accumulators (hi.hi / hi.lo / lo.hi) for intra-wave matrix ILP.
        // B 32x16 layout: lane col n = lane&15; elem j = E[n][32s + 8*half + j], elem 8+j = E[n][32s + 16 + 8*half + j].
        v8f acc0 = {0.f, 0.f, 0.f, 0.f, 0.f, 0.f, 0.f, 0.f};
        v8f acc1 = {0.f, 0.f, 0.f, 0.f, 0.f, 0.f, 0.f, 0.f};
        v8f acc2 = {0.f, 0.f, 0.f, 0.f, 0.f, 0.f, 0.f, 0.f};
        const unsigned short* bb = Bcur + l15 * BSTR + 8 * half;
#pragma unroll
        for (int s = 0; s < KSTEPS; ++s) {
            BfFrag bhi, blo;
            bhi.q[0] = *(const v4u*)(bb + 32 * s);
            bhi.q[1] = *(const v4u*)(bb + 32 * s + 16);
            blo.q[0] = *(const v4u*)(bb + BPLANE + 32 * s);
            blo.q[1] = *(const v4u*)(bb + BPLANE + 32 * s + 16);
            acc0 = __builtin_amdgcn_wmma_f32_16x16x32_bf16(
                false, fahi[s].v, false, bhi.v, (short)0, acc0, false, false);
            acc1 = __builtin_amdgcn_wmma_f32_16x16x32_bf16(
                false, fahi[s].v, false, blo.v, (short)0, acc1, false, false);
            acc2 = __builtin_amdgcn_wmma_f32_16x16x32_bf16(
                false, falo[s].v, false, bhi.v, (short)0, acc2, false, false);
        }

        // C layout: lane l, vgpr r -> row = r + 8*(l>>4), code col = l&15.
        const int   code = kt * 16 + l15;
        const float en   = enorm[code];
#pragma unroll
        for (int r = 0; r < 8; ++r) {
            float sc   = acc0[r] + (acc1[r] + acc2[r]);  // big term last
            float dcur = en - 2.0f * sc;
            if (dcur < best[r]) { best[r] = dcur; bidx[r] = code; }
        }
#if VQ_ASYNC_STAGE
        __builtin_amdgcn_s_wait_asynccnt(0);
#endif
        __syncthreads();   // Bnxt fully written & Bcur reads done before swap
    }

    // ---- Reduce across the 16 lanes of each half (tie: lower index) ----
#pragma unroll
    for (int r = 0; r < 8; ++r) {
        float b  = best[r];
        int   bi = bidx[r];
#pragma unroll
        for (int off = 8; off >= 1; off >>= 1) {
            float ob  = __shfl_xor(b, off, 32);
            int   obi = __shfl_xor(bi, off, 32);
            if (ob < b || (ob == b && obi < bi)) { b = ob; bi = obi; }
        }
        if (l15 == 0)
            idx_out[block_row + wave * ROWS_PER_WAVE + r + 8 * half] = bi;
    }
}

// ---------------------------------------------------------------------------
// Kernel 2: gather codebook rows -> output; per-block sum of (x - e)^2.
// Forward value of the straight-through output is exactly `quantized`.
// ---------------------------------------------------------------------------
__global__ __launch_bounds__(256) void vq_gather_kernel(
    const float* __restrict__ lat, const float* __restrict__ emb,
    const int* __restrict__ idx, float* __restrict__ out,
    float* __restrict__ partials)
{
    __shared__ float red[256];
    const int tid  = threadIdx.x;
    const int row  = blockIdx.x * 16 + (tid >> 4);
    const int d0   = (tid & 15) * 16;
    const int code = idx[row];

    const v4f* lp = (const v4f*)(lat + (size_t)row * VQ_D + d0);
    const v4f* ep = (const v4f*)(emb + (size_t)code * VQ_D + d0);
    v4f*       op = (v4f*)(out + (size_t)row * VQ_D + d0);

    float s = 0.f;
#pragma unroll
    for (int i = 0; i < 4; ++i) {
        v4f e = ep[i];
        v4f x = lp[i];
        op[i] = e;
        v4f df = x - e;
        s += df.x * df.x + df.y * df.y + df.z * df.z + df.w * df.w;
    }
    red[tid] = s;
    __syncthreads();
#pragma unroll
    for (int st = 128; st > 0; st >>= 1) {
        if (tid < st) red[tid] += red[tid + st];
        __syncthreads();
    }
    if (tid == 0) partials[blockIdx.x] = red[0];
}

// ---------------------------------------------------------------------------
// Kernel 3: deterministic fixed-order sum of partials; vq_loss scalar.
// vq_loss = (1 + BETA) * mean((lat - quant)^2),  BETA = 0.25
// ---------------------------------------------------------------------------
__global__ __launch_bounds__(256) void vq_loss_kernel(
    const float* __restrict__ partials, int nparts, float* __restrict__ loss_out)
{
    __shared__ float red[256];
    float s = 0.f;
    for (int i = threadIdx.x; i < nparts; i += 256) s += partials[i];
    red[threadIdx.x] = s;
    __syncthreads();
#pragma unroll
    for (int st = 128; st > 0; st >>= 1) {
        if (threadIdx.x < st) red[threadIdx.x] += red[threadIdx.x + st];
        __syncthreads();
    }
    if (threadIdx.x == 0)
        loss_out[0] = 1.25f * red[0] / ((float)VQ_N * (float)VQ_D);
}

// ---------------------------------------------------------------------------
extern "C" void kernel_launch(void* const* d_in, const int* in_sizes, int n_in,
                              void* d_out, int out_size, void* d_ws, size_t ws_size,
                              hipStream_t stream)
{
    const float* lat = (const float*)d_in[0];   // [8192, 2048] f32
    const float* emb = (const float*)d_in[1];   // [1024, 256]  f32
    float* out = (float*)d_out;                 // [N*256] quantized + [1] loss

    // Workspace layout (16B-aligned sections)
    const size_t plane_bytes = (size_t)VQ_K * VQ_D * sizeof(unsigned short); // 512KB
    unsigned short* ehi   = (unsigned short*)d_ws;
    unsigned short* elo   = (unsigned short*)((char*)d_ws + plane_bytes);
    float* enorm          = (float*)((char*)d_ws + 2 * plane_bytes);
    int*   indices        = (int*)((char*)d_ws + 2 * plane_bytes
                                              + VQ_K * sizeof(float));
    float* partials       = (float*)((char*)d_ws + 2 * plane_bytes
                                              + VQ_K * sizeof(float)
                                              + VQ_N * sizeof(int));

    vq_prep_kernel<<<VQ_K, 256, 0, stream>>>(emb, ehi, elo, enorm);

    // LDS: max(A staging 128*260*4 B, 2 B-buffers) = 133120 B -> 2 blocks/WGP
    size_t lds_bytes = (size_t)ROWS_PER_BLOCK * AS * sizeof(float);
    vq_argmin_kernel<<<VQ_N / ROWS_PER_BLOCK, 256, lds_bytes, stream>>>(
        lat, ehi, elo, enorm, indices);

    const int ngather = VQ_N / 16;  // 4096 blocks, 16 rows each
    vq_gather_kernel<<<ngather, 256, 0, stream>>>(lat, emb, indices, out, partials);

    vq_loss_kernel<<<1, 256, 0, stream>>>(partials, ngather,
                                          out + (size_t)VQ_N * VQ_D);
}
